// Block_71554155151896
// MI455X (gfx1250) — matrix operand order repaired and checked by reading, not compiled, not run on hardware
//
#include <hip/hip_runtime.h>
#include <hip/hip_bf16.h>

typedef __bf16 bf16;
typedef __attribute__((ext_vector_type(16))) __bf16 v16bf;
typedef __attribute__((ext_vector_type(8)))  float  v8f;
typedef __attribute__((ext_vector_type(4)))  unsigned int v4u;
typedef __attribute__((ext_vector_type(8)))  unsigned int v8u;

#define DEV static __device__ __forceinline__

DEV float silu_f(float x){ return x / (1.0f + __expf(-x)); }

// ---- WMMA fragment helpers (layouts per CDNA5 ISA §7.12.2) ----
// A: 16x32 bf16 from LDS, row-major [16][lda]; two 16B ds_load_b128 per lane.
DEV v16bf load_a_frag(const bf16* A, int lda, int k0, int lane){
  const int m  = lane & 15;
  const int kh = (lane >> 4) << 3;
  const bf16* p = A + (size_t)m * lda + k0 + kh;
  union { v4u u[2]; v16bf v; } t;
  t.u[0] = *(const v4u*)p;        // K = k0+kh+0..7
  t.u[1] = *(const v4u*)(p + 16); // K = k0+kh+16..23
  return t.v;
}
// B: pre-swizzled fragment-major: frag = 32 lanes x 16 bf16 contiguous (512 elems).
// One 32B load per lane -> two global_load_b128, zero shuffles.
DEV v16bf load_b_packed(const bf16* Bp, int lane){
  union { v8u u; v16bf v; } t;
  t.u = *(const v8u*)(Bp + (size_t)lane * 16);
  return t.v;
}
DEV v8f wmma_bf(v16bf a, v16bf b, v8f c){
  return __builtin_amdgcn_wmma_f32_16x16x32_bf16(false, a, false, b, (short)0, c,
                                                 false, false);
}

// C(16 x NT*16 fp32, LDS) = A(16xK bf16, LDS) * Bp(packed) [+bias]; one wave.
// A fragment loaded once per kt; NT accumulators live in VGPRs.
template<int NT>
DEV void gemm_p(const bf16* A, int lda, const bf16* __restrict__ Bp, int Ktiles,
                float* C, int ldc, const float* bias, int lane){
  v8f acc[NT] = {};
  for (int kt = 0; kt < Ktiles; ++kt){
    v16bf a = load_a_frag(A, lda, kt << 5, lane);
#pragma unroll
    for (int j = 0; j < NT; ++j)
      acc[j] = wmma_bf(a, load_b_packed(Bp + ((size_t)j * Ktiles + kt) * 512, lane),
                       acc[j]);
  }
  const int n = lane & 15, mo = (lane >> 4) << 3;
#pragma unroll
  for (int j = 0; j < NT; ++j){
    float bv = bias ? bias[j * 16 + n] : 0.0f;
#pragma unroll
    for (int r = 0; r < 8; ++r)
      C[(size_t)(mo + r) * ldc + j * 16 + n] = acc[j][r] + bv;
  }
}

// per-row LayerNorm(eps 1e-6)*g+be then SiLU; rewrites fp32, emits bf16 copy.
DEV void ln_silu_16x64(float* X, const float* g, const float* be, bf16* Xb, int lane){
  if (lane < 16){
    float* row = X + lane * 64;
    float mu = 0.f;
    for (int i = 0; i < 64; ++i) mu += row[i];
    mu *= (1.0f / 64.0f);
    float var = 0.f;
    for (int i = 0; i < 64; ++i){ float d = row[i] - mu; var += d * d; }
    var *= (1.0f / 64.0f);
    float rs = rsqrtf(var + 1e-6f);
    for (int i = 0; i < 64; ++i){
      float y = silu_f((row[i] - mu) * rs * g[i] + be[i]);
      row[i] = y;
      Xb[lane * 64 + i] = (bf16)y;
    }
  }
  __syncthreads();
}

DEV void store4_bf(bf16* p, float4 v){
  p[0] = (bf16)v.x; p[1] = (bf16)v.y; p[2] = (bf16)v.z; p[3] = (bf16)v.w;
}

// ---------------- prep kernels ----------------
// Pack W[K][Nsrc] fp32 -> fragment-major bf16: out[nt][kt][lane][i], zero-padding
// columns Nsrc..Ntiles*16-1 (absorbs alpha-W3 4->16 pad).
__global__ void k_pack(const float* __restrict__ W, bf16* __restrict__ out,
                       int K, int Nsrc, int Ntiles){
  const int Ktiles = K >> 5;
  const size_t total = (size_t)Ntiles * Ktiles * 512;
  for (size_t idx = (size_t)blockIdx.x * blockDim.x + threadIdx.x; idx < total;
       idx += (size_t)gridDim.x * blockDim.x){
    int i    = (int)(idx & 15);
    int lane = (int)((idx >> 4) & 31);
    size_t rest = idx >> 9;
    int kt = (int)(rest % Ktiles);
    int nt = (int)(rest / Ktiles);
    int n = nt * 16 + (lane & 15);
    int k = kt * 32 + ((lane >> 4) << 3) + (i & 7) + ((i >> 3) << 4);
    float v = (n < Nsrc) ? W[(size_t)k * Nsrc + n] : 0.0f;
    out[idx] = (bf16)v;
  }
}
__global__ void k_fill_f32(float* d, float v, size_t n){
  size_t i = (size_t)blockIdx.x * blockDim.x + threadIdx.x;
  for (; i < n; i += (size_t)gridDim.x * blockDim.x) d[i] = v;
}
__global__ void k_fill_i32(int* d, int v, size_t n){
  size_t i = (size_t)blockIdx.x * blockDim.x + threadIdx.x;
  for (; i < n; i += (size_t)gridDim.x * blockDim.x) d[i] = v;
}

// ---------------- alpha MLP + segment max ----------------
__global__ void __launch_bounds__(32)
k_alpha(const float* __restrict__ elen, const int* __restrict__ edst,
        const bf16* __restrict__ W1, const float* b1, const float* g1, const float* be1,
        const bf16* __restrict__ W2, const float* b2, const float* g2, const float* be2,
        const bf16* __restrict__ W3p, const float* b3,
        float* __restrict__ logits, int* __restrict__ mx, int E){
  __shared__ bf16  Xb[16 * 64];
  __shared__ float Of[16 * 64];
  const int lane = threadIdx.x;
  const int e0 = blockIdx.x << 4;
  const float4* el4 = (const float4*)elen;
  for (int t = lane; t < 16 * 16; t += 32){
    int r = t >> 4, c4 = (t & 15) << 2;
    int e = e0 + r; if (e >= E) e = E - 1;
    store4_bf(&Xb[r * 64 + c4], el4[(size_t)e * 16 + (c4 >> 2)]);
  }
  __syncthreads();
  gemm_p<4>(Xb, 64, W1, 2, Of, 64, b1, lane);
  __syncthreads();
  ln_silu_16x64(Of, g1, be1, Xb, lane);
  gemm_p<4>(Xb, 64, W2, 2, Of, 64, b2, lane);
  __syncthreads();
  ln_silu_16x64(Of, g2, be2, Xb, lane);
  v8f acc = {};
  for (int kt = 0; kt < 2; ++kt)
    acc = wmma_bf(load_a_frag(Xb, 64, kt << 5, lane),
                  load_b_packed(W3p + (size_t)kt * 512, lane), acc);
  const int n = lane & 15, mo = (lane >> 4) << 3;
  if (n < 4){
#pragma unroll
    for (int r = 0; r < 8; ++r){
      int e = e0 + mo + r;
      if (e < E){
        float v = acc[r] + b3[n];
        logits[(size_t)e * 4 + n] = v;
        int bi = __float_as_int(v);
        int enc = (bi >= 0) ? bi : (bi ^ 0x7fffffff);   // order-preserving float->int
        atomicMax(&mx[edst[e] * 4 + n], enc);
      }
    }
  }
}

// ---------------- exp + segment denominator ----------------
__global__ void k_expden(float* __restrict__ lg /* in-place -> ex */,
                         const int* __restrict__ edst, const int* __restrict__ mx,
                         float* __restrict__ den, int EH){
  int i = blockIdx.x * blockDim.x + threadIdx.x;
  if (i >= EH) return;
  int e = i >> 2, h = i & 3;
  int enc = mx[edst[e] * 4 + h];
  int bi = (enc >= 0) ? enc : (enc ^ 0x7fffffff);
  float m = __int_as_float(bi);
  if (!(m > -INFINITY && m < INFINITY)) m = 0.0f;   // where(isfinite(mx), mx, 0)
  float ex = __expf(lg[i] - m);
  lg[i] = ex;
  atomicAdd(&den[edst[e] * 4 + h], ex);
}

// ---------------- main fused edge kernel ----------------
__global__ void __launch_bounds__(32)
k_edge(const float* __restrict__ node_in, const float* __restrict__ edge_sh,
       const float* __restrict__ elen,
       const int* __restrict__ esrc, const int* __restrict__ edst,
       const bf16* __restrict__ Wpre,
       const bf16* __restrict__ Wr1, const float* br1,
       const bf16* __restrict__ Wr2, const float* br2,
       const bf16* __restrict__ Wtp, const bf16* __restrict__ Wedge,
       const bf16* __restrict__ sW1, const float* sb1, const float* sg1, const float* sbe1,
       const bf16* __restrict__ sW2, const float* sb2, const float* sg2, const float* sbe2,
       const bf16* __restrict__ sW3, const float* sb3,
       const float* __restrict__ ex, const float* __restrict__ den,
       float* __restrict__ nfea, float* __restrict__ edge_scalar, int E){
  __shared__ bf16  inA[16 * 192];    // [src(64)|dst(64)|len(64)] bf16
  __shared__ bf16  tpA[16 * 576];    // msg (x) sh operand; later reused as eout bf16
  __shared__ bf16  hB [16 * 64];
  __shared__ float shv[16 * 12];
  __shared__ float msgF[16 * 64];
  __shared__ float f0[16 * 64];
  __shared__ float f1[16 * 64];
  __shared__ float eout[16 * 256];
  __shared__ float alph[16 * 4];
  __shared__ int   dstid[16];

  const int lane = threadIdx.x;
  const int e0 = blockIdx.x << 4;
  if (lane == 0) __builtin_prefetch(Wtp, 0, 3);   // global_prefetch_b8 into L2

  const float4* ni4 = (const float4*)node_in;
  const float4* el4 = (const float4*)elen;
  for (int t = lane; t < 16 * 16; t += 32){
    int r = t >> 4, q = t & 15, c4 = q << 2;
    int e = e0 + r; if (e >= E) e = E - 1;
    int s = esrc[e], d = edst[e];
    store4_bf(&inA[r * 192 + c4],       ni4[(size_t)s * 16 + q]);
    store4_bf(&inA[r * 192 + 64 + c4],  ni4[(size_t)d * 16 + q]);
    store4_bf(&inA[r * 192 + 128 + c4], el4[(size_t)e * 16 + q]);
  }
  for (int t = lane; t < 16 * 9; t += 32){
    int r = t / 9, c = t - r * 9;
    int e = e0 + r; if (e >= E) e = E - 1;
    shv[r * 12 + c] = edge_sh[(size_t)e * 9 + c];
  }
  for (int t = lane; t < 16 * 4; t += 32){
    int r = t >> 2, h = t & 3;
    int e = e0 + r; if (e >= E) e = E - 1;
    alph[t] = ex[(size_t)e * 4 + h] / (den[edst[e] * 4 + h] + 1e-12f);
  }
  if (lane < 16){ int e = e0 + lane; dstid[lane] = (e < E) ? edst[e] : -1; }
  __syncthreads();

  // message = [src|dst|len] @ W_pre   (16x192 * 192x64)
  gemm_p<4>(inA, 192, Wpre, 6, msgF, 64, nullptr, lane);
  __syncthreads();

  // tp operand: tpA[e, f*9+s] = msg[e,f] * sh[e,s]   (K = 576, exactly 18 k-tiles)
  for (int t = lane; t < 16 * 576; t += 32){
    int r = t / 576, c = t - r * 576;
    int f = c / 9, s = c - f * 9;
    tpA[t] = (bf16)(msgF[r * 64 + f] * shv[r * 12 + s]);
  }
  // radial hidden: silu(len @ W_r1 + b_r1)
  gemm_p<4>(inA + 128, 192, Wr1, 2, f0, 64, br1, lane);
  __syncthreads();
  for (int t = lane; t < 16 * 64; t += 32) hB[t] = (bf16)silu_f(f0[t]);
  __syncthreads();

  // per head: radial chunk, tp chunk, value * radial * alpha
  for (int h = 0; h < 4; ++h){
    gemm_p<4>(hB, 64, Wr2 + (size_t)h * 4 * 2 * 512, 2, f0, 64, br2 + h * 64, lane);
    gemm_p<4>(tpA, 576, Wtp + (size_t)h * 4 * 18 * 512, 18, f1, 64, nullptr, lane);
    __syncthreads();
    for (int t = lane; t < 16 * 64; t += 32){
      int r = t >> 6, c = t & 63;
      eout[r * 256 + h * 64 + c] = f1[t] * f0[t] * alph[r * 4 + h];
    }
    __syncthreads();
  }

  // scatter-aggregate messages per node
  for (int t = lane; t < 16 * 256; t += 32){
    int r = t >> 8, c = t & 255;
    int d = dstid[r];
    if (d >= 0) atomicAdd(&nfea[(size_t)d * 256 + c], eout[t]);
  }

  // edge scalar head: (eout @ W_edge) -> mlp3 -> [16,32]
  bf16* eoutB = tpA;                              // reuse (16*256 <= 16*576)
  for (int t = lane; t < 16 * 256; t += 32) eoutB[t] = (bf16)eout[t];
  __syncthreads();
  gemm_p<4>(eoutB, 256, Wedge, 8, f1, 64, nullptr, lane);
  __syncthreads();
  for (int t = lane; t < 16 * 64; t += 32) hB[t] = (bf16)f1[t];
  __syncthreads();
  gemm_p<4>(hB, 64, sW1, 2, f0, 64, sb1, lane);
  __syncthreads();
  ln_silu_16x64(f0, sg1, sbe1, hB, lane);
  gemm_p<4>(hB, 64, sW2, 2, f1, 64, sb2, lane);
  __syncthreads();
  ln_silu_16x64(f1, sg2, sbe2, hB, lane);
  const int n = lane & 15, mo = (lane >> 4) << 3;
  for (int j = 0; j < 2; ++j){
    v8f acc = {};
    for (int kt = 0; kt < 2; ++kt)
      acc = wmma_bf(load_a_frag(hB, 64, kt << 5, lane),
                    load_b_packed(sW3 + ((size_t)j * 2 + kt) * 512, lane), acc);
#pragma unroll
    for (int r = 0; r < 8; ++r){
      int e = e0 + mo + r;
      if (e < E) edge_scalar[(size_t)e * 32 + j * 16 + n] = acc[r] + sb3[j * 16 + n];
    }
  }
}

// ---------------- node output GEMM ----------------
__global__ void __launch_bounds__(32)
k_nodeout(const float* __restrict__ nfea, const bf16* __restrict__ Wout,
          float* __restrict__ out, int Nn){
  __shared__ bf16 A[16 * 256];
  const int lane = threadIdx.x;
  const int m0 = blockIdx.x << 4;
  const float4* nf4 = (const float4*)nfea;
  for (int t = lane; t < 16 * 64; t += 32){
    int r = t >> 6, q = t & 63;
    int nd = m0 + r; if (nd >= Nn) nd = Nn - 1;
    store4_bf(&A[r * 256 + (q << 2)], nf4[(size_t)nd * 64 + q]);
  }
  __syncthreads();
  v8f acc[4] = {};
  for (int kt = 0; kt < 8; ++kt){
    v16bf a = load_a_frag(A, 256, kt << 5, lane);
#pragma unroll
    for (int j = 0; j < 4; ++j)
      acc[j] = wmma_bf(a, load_b_packed(Wout + ((size_t)j * 8 + kt) * 512, lane), acc[j]);
  }
  const int n = lane & 15, mo = (lane >> 4) << 3;
#pragma unroll
  for (int j = 0; j < 4; ++j){
#pragma unroll
    for (int r = 0; r < 8; ++r){
      int nd = m0 + mo + r;
      if (nd < Nn) out[(size_t)nd * 64 + j * 16 + n] = acc[j][r];
    }
  }
}

// ---------------- host launcher ----------------
extern "C" void kernel_launch(void* const* d_in, const int* in_sizes, int n_in,
                              void* d_out, int out_size, void* d_ws, size_t ws_size,
                              hipStream_t stream){
  (void)n_in; (void)out_size; (void)ws_size;
  const float* node_in = (const float*)d_in[0];
  const float* edge_sh = (const float*)d_in[2];
  const float* elen    = (const float*)d_in[3];
  const int*   esrc    = (const int*)d_in[4];
  const int*   edst    = (const int*)d_in[5];
  const float* W_pre = (const float*)d_in[7];
  const float* W_r1  = (const float*)d_in[8];
  const float* b_r1  = (const float*)d_in[9];
  const float* W_r2  = (const float*)d_in[10];
  const float* b_r2  = (const float*)d_in[11];
  const float* W_tp  = (const float*)d_in[12];
  const float* aW1  = (const float*)d_in[13];
  const float* ab1  = (const float*)d_in[14];
  const float* ag1  = (const float*)d_in[15];
  const float* abe1 = (const float*)d_in[16];
  const float* aW2  = (const float*)d_in[17];
  const float* ab2  = (const float*)d_in[18];
  const float* ag2  = (const float*)d_in[19];
  const float* abe2 = (const float*)d_in[20];
  const float* aW3  = (const float*)d_in[21];
  const float* ab3  = (const float*)d_in[22];
  const float* W_edge = (const float*)d_in[23];
  const float* sW1  = (const float*)d_in[24];
  const float* sb1  = (const float*)d_in[25];
  const float* sg1  = (const float*)d_in[26];
  const float* sbe1 = (const float*)d_in[27];
  const float* sW2  = (const float*)d_in[28];
  const float* sb2  = (const float*)d_in[29];
  const float* sg2  = (const float*)d_in[30];
  const float* sbe2 = (const float*)d_in[31];
  const float* sW3  = (const float*)d_in[32];
  const float* sb3  = (const float*)d_in[33];
  const float* W_out = (const float*)d_in[34];

  const int N = in_sizes[0] / 64;
  const int E = in_sizes[4];

  // workspace bump allocator (256B aligned)
  char* p = (char*)d_ws;
  auto alloc = [&](size_t bytes) -> void* {
    void* r = (void*)p; p += (bytes + 255) & ~(size_t)255; return r;
  };
  auto palloc = [&](int Ktiles, int Ntiles) -> bf16* {
    return (bf16*)alloc((size_t)Ktiles * Ntiles * 512 * sizeof(bf16));
  };
  bf16* Wpre_b  = palloc(6, 4);
  bf16* Wr1_b   = palloc(2, 4);
  bf16* Wr2_b   = palloc(2, 16);
  bf16* Wtp_b   = palloc(18, 16);
  bf16* aW1_b   = palloc(2, 4);
  bf16* aW2_b   = palloc(2, 4);
  bf16* aW3p_b  = palloc(2, 1);     // 64x4 zero-padded to 64x16
  bf16* Wedge_b = palloc(8, 4);
  bf16* sW1_b   = palloc(2, 4);
  bf16* sW2_b   = palloc(2, 4);
  bf16* sW3_b   = palloc(2, 2);
  bf16* Wout_b  = palloc(8, 4);
  float* logits = (float*)alloc((size_t)E * 4 * sizeof(float));  // then ex in-place
  int*   mx     = (int*)  alloc((size_t)N * 4 * sizeof(int));
  float* den    = (float*)alloc((size_t)N * 4 * sizeof(float));
  float* nfea   = (float*)alloc((size_t)N * 256 * sizeof(float));

  auto pack = [&](const float* W, bf16* d, int K, int Nsrc, int Ntiles){
    int total = (K >> 5) * Ntiles * 512;
    k_pack<<<(total + 255) / 256, 256, 0, stream>>>(W, d, K, Nsrc, Ntiles);
  };
  pack(W_pre,  Wpre_b,  192, 64, 4);
  pack(W_r1,   Wr1_b,    64, 64, 4);
  pack(W_r2,   Wr2_b,    64, 256, 16);
  pack(W_tp,   Wtp_b,   576, 256, 16);
  pack(aW1,    aW1_b,    64, 64, 4);
  pack(aW2,    aW2_b,    64, 64, 4);
  pack(aW3,    aW3p_b,   64, 4, 1);
  pack(W_edge, Wedge_b, 256, 64, 4);
  pack(sW1,    sW1_b,    64, 64, 4);
  pack(sW2,    sW2_b,    64, 64, 4);
  pack(sW3,    sW3_b,    64, 32, 2);
  pack(W_out,  Wout_b,  256, 64, 4);

  k_fill_i32<<<1024, 256, 0, stream>>>(mx, (int)0x80000000, (size_t)N * 4);
  k_fill_f32<<<1024, 256, 0, stream>>>(den, 0.0f, (size_t)N * 4);
  k_fill_f32<<<2048, 256, 0, stream>>>(nfea, 0.0f, (size_t)N * 256);

  const int etiles = (E + 15) / 16;
  k_alpha<<<etiles, 32, 0, stream>>>(elen, edst,
      aW1_b, ab1, ag1, abe1, aW2_b, ab2, ag2, abe2, aW3p_b, ab3,
      logits, mx, E);
  k_expden<<<((E * 4) + 255) / 256, 256, 0, stream>>>(logits, edst, mx, den, E * 4);

  float* node_out    = (float*)d_out;
  float* edge_scalar = node_out + (size_t)N * 64;
  k_edge<<<etiles, 32, 0, stream>>>(node_in, edge_sh, elen, esrc, edst,
      Wpre_b, Wr1_b, b_r1, Wr2_b, b_r2, Wtp_b, Wedge_b,
      sW1_b, sb1, sg1, sbe1, sW2_b, sb2, sg2, sbe2, sW3_b, sb3,
      logits /* ex */, den, nfea, edge_scalar, E);

  k_nodeout<<<(N + 15) / 16, 32, 0, stream>>>(nfea, Wout_b, node_out, N);
}
